// MultiSimilarityLoss_48515950576321
// MI455X (gfx1250) — compile-verified, matrix operand order checked by reading, not run on hardware
//
#include <hip/hip_runtime.h>
#include <math.h>

// ---------------------------------------------------------------------------
// MultiSimilarity loss, fused for MI455X (gfx1250):
//   sim = normalize(feats) @ normalize(feats)^T  via v_wmma_f32_16x16x32_f16
//   16-row sim strip (16 x 4096 f32 = 256KB) lives in WGP LDS (320KB budget),
//   three dependent mining passes run from LDS with wave32 shuffle reductions.
// ---------------------------------------------------------------------------

typedef __attribute__((ext_vector_type(16))) _Float16 v16h;
typedef __attribute__((ext_vector_type(8)))  _Float16 v8h;
typedef __attribute__((ext_vector_type(8)))  float    v8f;

#define NB      4096
#define DIM     512
#define NCLS    100
#define THRESH    0.5f
#define MARGIN    0.1f
#define SCALE_POS 2.0f
#define SCALE_NEG 40.0f
#define EPSILON   1e-5f
#define BIGF      3.0e38f

// ------------------------- prep: normalize + label index -------------------
__global__ void __launch_bounds__(256)
ms_prep_kernel(const float* __restrict__ feats,
               const float* __restrict__ labels,
               _Float16* __restrict__ fnorm,
               int* __restrict__ lab) {
  const int wave = (blockIdx.x * blockDim.x + threadIdx.x) >> 5;  // one wave32 per row
  const int lane = threadIdx.x & 31;
  if (wave >= NB) return;

  const float* row = feats + (size_t)wave * DIM;
  float ss = 0.0f;
  for (int k = lane; k < DIM; k += 32) { float x = row[k]; ss += x * x; }
  #pragma unroll
  for (int off = 16; off > 0; off >>= 1) ss += __shfl_xor(ss, off, 32);
  const float inv = rsqrtf(fmaxf(ss, 1e-30f));

  _Float16* orow = fnorm + (size_t)wave * DIM;
  for (int k = lane; k < DIM; k += 32) orow[k] = (_Float16)(row[k] * inv);

  // one-hot -> class index
  const float* lrow = labels + (size_t)wave * NCLS;
  int li = 0;
  for (int c = lane; c < NCLS; c += 32) if (lrow[c] > 0.5f) li = c;
  #pragma unroll
  for (int off = 16; off > 0; off >>= 1) li = max(li, __shfl_xor(li, off, 32));
  if (lane == 0) lab[wave] = li;
}

// --------------- fused WMMA GEMM strip + multi-similarity mining -----------
__global__ void __launch_bounds__(256)
ms_main_kernel(const _Float16* __restrict__ fnorm,
               const int* __restrict__ lab,
               float* __restrict__ partials) {
  extern __shared__ char smem[];
  float* sim  = (float*)smem;                              // [16][4096] = 256KB
  int*   labl = (int*)(smem + (size_t)16 * NB * sizeof(float)); // [4096] = 16KB
  __shared__ float wloss[8];

  const int tid  = threadIdx.x;
  const int w    = tid >> 5;        // wave id 0..7
  const int lane = tid & 31;
  const int half = lane >> 4;       // 0: lanes 0-15, 1: lanes 16-31
  const int l15  = lane & 15;
  const int i0   = blockIdx.x * 16; // first global row of this strip

  // stage labels into LDS (reused heavily by the mining passes)
  for (int j = tid; j < NB; j += 256) labl[j] = lab[j];

  // ---- preload all A fragments (16 rows x K=512) into registers ----
  // Per ISA 16-bit A layout: lane (l&15) = row M; lanes 0-15 hold K = base..base+7
  // in VGPR0-3 and K = 16+base..16+base+7 in VGPR4-7 with base = 8*half.
  const _Float16* arow = fnorm + (size_t)(i0 + l15) * DIM + half * 8;
  v16h afrag[16];
  #pragma unroll
  for (int ks = 0; ks < 16; ++ks) {
    v8h lo = *(const v8h*)(arow + ks * 32);
    v8h hi = *(const v8h*)(arow + ks * 32 + 16);
    afrag[ks] = __builtin_shufflevector(lo, hi,
        0, 1, 2, 3, 4, 5, 6, 7, 8, 9, 10, 11, 12, 13, 14, 15);
  }

  // ---- GEMM: 256 column tiles of 16, 8 waves cover 128 columns/iter ----
  for (int jt = 0; jt < NB; jt += 128) {
    const int j0 = jt + w * 16;
    const _Float16* brow = fnorm + (size_t)(j0 + l15) * DIM + half * 8;
    v8f c = {};
    #pragma unroll
    for (int ks = 0; ks < 16; ++ks) {
      v8h lo = *(const v8h*)(brow + ks * 32);
      v8h hi = *(const v8h*)(brow + ks * 32 + 16);
      v16h b = __builtin_shufflevector(lo, hi,
          0, 1, 2, 3, 4, 5, 6, 7, 8, 9, 10, 11, 12, 13, 14, 15);
      // (neg_a, A, neg_b, B, c_mod, C, reuse_a, reuse_b)
      c = __builtin_amdgcn_wmma_f32_16x16x32_f16(
          false, afrag[ks], false, b, (short)0, c, false, false);
    }
    // C/D layout: lane(0-15)=N col; VGPR v holds M=v (lanes 0-15) / M=8+v (16-31)
    #pragma unroll
    for (int v = 0; v < 8; ++v)
      sim[(size_t)(v + 8 * half) * NB + (j0 + l15)] = c[v];
  }
  __syncthreads();

  // ---- mining passes: wave w owns rows 2w and 2w+1, all shuffle-reduced ----
  float wl = 0.0f;
  #pragma unroll
  for (int rr = 0; rr < 2; ++rr) {
    const int r  = w * 2 + rr;
    const int gi = i0 + r;
    const int li = labl[gi];
    const float* srow = sim + (size_t)r * NB;

    // pass 1: hardest (minimum) positive similarity
    float pmin = BIGF;
    for (int j = lane; j < NB; j += 32) {
      float s = srow[j];
      if (labl[j] == li && j != gi && s < 1.0f - EPSILON) pmin = fminf(pmin, s);
    }
    #pragma unroll
    for (int off = 16; off > 0; off >>= 1) pmin = fminf(pmin, __shfl_xor(pmin, off, 32));

    // pass 2: kept negatives -> max and exp-sum
    float nmax = -BIGF, nsum = 0.0f;
    for (int j = lane; j < NB; j += 32) {
      float s = srow[j];
      if (labl[j] != li && (s + MARGIN > pmin)) {
        nmax = fmaxf(nmax, s);
        nsum += expf(SCALE_NEG * (s - THRESH));
      }
    }
    #pragma unroll
    for (int off = 16; off > 0; off >>= 1) {
      nmax = fmaxf(nmax, __shfl_xor(nmax, off, 32));
      nsum += __shfl_xor(nsum, off, 32);
    }

    // pass 3: kept positives -> exp-sum
    float psum = 0.0f;
    for (int j = lane; j < NB; j += 32) {
      float s = srow[j];
      if (labl[j] == li && j != gi && s < 1.0f - EPSILON && (s - MARGIN < nmax))
        psum += expf(-SCALE_POS * (s - THRESH));
    }
    #pragma unroll
    for (int off = 16; off > 0; off >>= 1) psum += __shfl_xor(psum, off, 32);

    if (lane == 0) {
      // exp sums are strictly positive iff any element was kept
      if (psum > 0.0f && nsum > 0.0f)
        wl += log1pf(psum) / SCALE_POS + log1pf(nsum) / SCALE_NEG;
    }
  }
  if (lane == 0) wloss[w] = wl;
  __syncthreads();
  if (tid == 0) {
    float t = 0.0f;
    #pragma unroll
    for (int k = 0; k < 8; ++k) t += wloss[k];
    partials[blockIdx.x] = t;
  }
}

// ------------------------- deterministic final reduce ----------------------
__global__ void __launch_bounds__(256)
ms_final_kernel(const float* __restrict__ partials, float* __restrict__ out) {
  __shared__ float red[256];
  const int t = threadIdx.x;
  red[t] = partials[t];   // exactly 256 partials (4096 / 16)
  __syncthreads();
  #pragma unroll
  for (int s = 128; s > 0; s >>= 1) {
    if (t < s) red[t] += red[t + s];
    __syncthreads();
  }
  if (t == 0) out[0] = red[0] / (float)NB;
}

// ---------------------------------------------------------------------------
extern "C" void kernel_launch(void* const* d_in, const int* in_sizes, int n_in,
                              void* d_out, int out_size, void* d_ws, size_t ws_size,
                              hipStream_t stream) {
  const float* feats  = (const float*)d_in[0];
  const float* labels = (const float*)d_in[1];

  char* ws = (char*)d_ws;
  _Float16* fnorm   = (_Float16*)ws;                                   // 4 MB
  int*      lab     = (int*)(ws + (size_t)NB * DIM * sizeof(_Float16)); // 16 KB
  float*    partials = (float*)(ws + (size_t)NB * DIM * sizeof(_Float16)
                                   + (size_t)NB * sizeof(int));         // 1 KB

  // 1) row L2-normalize to f16 + label index (one wave32 per row)
  ms_prep_kernel<<<NB / 8, 256, 0, stream>>>(feats, labels, fnorm, lab);

  // 2) fused WMMA strip GEMM + mining; 256 blocks of 16 rows.
  //    Dynamic LDS: 16x4096 f32 sim strip (256KB) + 4096 labels (16KB) < 320KB WGP LDS.
  const size_t lds_bytes = (size_t)16 * NB * sizeof(float) + (size_t)NB * sizeof(int);
  ms_main_kernel<<<NB / 16, 256, lds_bytes, stream>>>(fnorm, lab, partials);

  // 3) deterministic tree sum -> mean loss
  ms_final_kernel<<<1, 256, 0, stream>>>(partials, (float*)d_out);
}
